// RetNet_16767552323875
// MI455X (gfx1250) — matrix operand order repaired and checked by reading, not compiled, not run on hardware
//
#include <hip/hip_runtime.h>
#include <hip/hip_bf16.h>
#include <math.h>

// ---------------------------------------------------------------------------
// RetNet forward for MI455X (gfx1250, wave32, WMMA).
// All heavy math runs through v_wmma_f32_16x16x32_bf16 (bf16 in, fp32 acc).
// Retention is fused (flash-style): the 268MB score matrix never hits memory.
// GEMM: compile-time layout/epilogue, LDS ping-pong double buffering, pointer-
// increment addressing, contiguous ds_load_b128 fragments, global prefetch.
// ---------------------------------------------------------------------------

typedef __bf16 bf16;
typedef __attribute__((ext_vector_type(16))) __bf16 v16bf;
typedef __attribute__((ext_vector_type(8)))  __bf16 v8bf;
typedef __attribute__((ext_vector_type(8)))  float  v8f;
typedef __attribute__((ext_vector_type(4)))  unsigned int v4u;  // 16B

constexpr int kB = 4, kS = 2048, kH = 512, kHeads = 4, kHD = 128, kFFN = 2048, kLayers = 3;
constexpr float kEps = 1e-5f;

__device__ __forceinline__ v16bf cat8(v8bf a, v8bf b) {
  return __builtin_shufflevector(a, b, 0, 1, 2, 3, 4, 5, 6, 7, 8, 9, 10, 11, 12, 13, 14, 15);
}
__device__ __forceinline__ v16bf load_frag16(const bf16* p) {   // two ds_load_b128
  return cat8(*(const v8bf*)p, *(const v8bf*)(p + 8));
}

// ---------------------------------------------------------------------------
// Elementwise / packing kernels
// ---------------------------------------------------------------------------
__global__ void copy_f32_kernel(const float* __restrict__ in, float* __restrict__ out, size_t n) {
  size_t i = (size_t)blockIdx.x * blockDim.x + threadIdx.x;
  size_t st = (size_t)gridDim.x * blockDim.x;
  for (; i < n; i += st) out[i] = in[i];
}

__global__ void convert_f32_bf16_kernel(const float* __restrict__ in, bf16* __restrict__ out, size_t n) {
  size_t i = (size_t)blockIdx.x * blockDim.x + threadIdx.x;
  size_t st = (size_t)gridDim.x * blockDim.x;
  for (; i < n; i += st) out[i] = (bf16)in[i];
}

// WQ/WK/WV layout (n,h,d) -> packed (h, n*HD+d) so projections are one GEMM.
__global__ void pack_qkv_kernel(const float* __restrict__ in, bf16* __restrict__ out) {
  size_t i = (size_t)blockIdx.x * blockDim.x + threadIdx.x;
  const size_t total = (size_t)kHeads * kH * kHD;
  if (i >= total) return;
  int d = (int)(i % kHD);
  size_t r = i / kHD;
  int h = (int)(r % kH);
  int n = (int)(r / kH);
  out[(size_t)h * kH + (size_t)n * kHD + d] = (bf16)in[i];
}

// LayerNorm over H=512, one block (256 thr) per row, bf16 output.
__global__ __launch_bounds__(256) void layernorm_bf16_kernel(
    const float* __restrict__ X, const float* __restrict__ w, const float* __restrict__ b,
    bf16* __restrict__ out) {
  __shared__ float red[256];
  const int t = threadIdx.x;
  const float* xr = X + (size_t)blockIdx.x * kH;
  float x0 = xr[t], x1 = xr[t + 256];
  red[t] = x0 + x1;
  __syncthreads();
  for (int s = 128; s > 0; s >>= 1) { if (t < s) red[t] += red[t + s]; __syncthreads(); }
  float mean = red[0] * (1.0f / kH);
  __syncthreads();
  float d0 = x0 - mean, d1 = x1 - mean;
  red[t] = d0 * d0 + d1 * d1;
  __syncthreads();
  for (int s = 128; s > 0; s >>= 1) { if (t < s) red[t] += red[t + s]; __syncthreads(); }
  float inv = rsqrtf(red[0] * (1.0f / kH) + kEps);
  bf16* orow = out + (size_t)blockIdx.x * kH;
  orow[t]       = (bf16)(d0 * inv * w[t]       + b[t]);
  orow[t + 256] = (bf16)(d1 * inv * w[t + 256] + b[t + 256]);
}

// xPos rotary on Q (upscale) and K (downscale), fp32 in -> bf16 out, same layout.
__global__ void xpos_kernel(const float* __restrict__ Qf, const float* __restrict__ Kf,
                            bf16* __restrict__ Qh, bf16* __restrict__ Kh) {
  const int half = kHD / 2;
  size_t idx = (size_t)blockIdx.x * blockDim.x + threadIdx.x;
  const size_t total = (size_t)kB * kS * kHeads * half;
  if (idx >= total) return;
  int j = (int)(idx % half);
  size_t r = idx / half;
  int n = (int)(r % kHeads); r /= kHeads;
  int s = (int)(r % kS);
  int b = (int)(r / kS);
  float sj    = ((float)j + 0.4f * kHD) / (1.4f * kHD);
  float scale = __powf(sj, (float)s / 512.0f);
  float invf  = __powf(10000.0f, -(float)j / (float)half);
  float ang   = (float)s * invf;
  float sn, cs;
  __sincosf(ang, &sn, &cs);
  size_t base = ((size_t)b * kS + s) * kH + (size_t)n * kHD + 2 * j;
  float q1 = Qf[base], q2 = Qf[base + 1];
  Qh[base]     = (bf16)((q1 * cs - q2 * sn) * scale);
  Qh[base + 1] = (bf16)((q2 * cs + q1 * sn) * scale);
  float is = 1.0f / scale;
  float k1 = Kf[base], k2 = Kf[base + 1];
  Kh[base]     = (bf16)((k1 * cs - k2 * sn) * is);
  Kh[base + 1] = (bf16)((k2 * cs + k1 * sn) * is);
}

// GroupNorm over HD per (b,s,head) + gn affine + SiLU(G) gate -> bf16.
__global__ __launch_bounds__(128) void gate_kernel(
    const float* __restrict__ Yf, const float* __restrict__ Gf,
    const float* __restrict__ gnw, const float* __restrict__ gnb, bf16* __restrict__ out) {
  __shared__ float red[128];
  const int t = threadIdx.x;
  const int n = blockIdx.x % kHeads;
  const size_t srow = blockIdx.x / kHeads;          // (b*S + s)
  const size_t base = srow * kH + (size_t)n * kHD;
  float y = Yf[base + t];
  red[t] = y;
  __syncthreads();
  for (int s = 64; s > 0; s >>= 1) { if (t < s) red[t] += red[t + s]; __syncthreads(); }
  float m = red[0] * (1.0f / kHD);
  __syncthreads();
  float d = y - m;
  red[t] = d * d;
  __syncthreads();
  for (int s = 64; s > 0; s >>= 1) { if (t < s) red[t] += red[t + s]; __syncthreads(); }
  float inv = rsqrtf(red[0] * (1.0f / kHD) + kEps);
  int h = n * kHD + t;
  float yn = d * inv * gnw[h] + gnb[h];
  float g = Gf[base + t];
  float sig = 1.0f / (1.0f + __expf(-g));
  out[base + t] = (bf16)(g * sig * yn);
}

// ---------------------------------------------------------------------------
// WMMA GEMM: C[M,N] = A[M,K](bf16) * B (bf16; TRANSB: stored N x K).
// Workgroup tile 64x64, 8 waves; each wave: one A-frag x two B-frags = 2 WMMA
// per K-step.  B tile kept TRANSPOSED in LDS (Bs[n][k]) so all fragment reads
// are contiguous ds_load_b128.  LDS ping-pong double buffering: next tile's
// global loads are issued before the WMMAs, stored after -> 1 barrier/step.
//   EPI 0: fp32 out    1: +bias, exact GELU, bf16 out
//   EPI 2: +bias+resid fp32 out   3: +resid fp32 out
// ---------------------------------------------------------------------------
template <int TRANSB, int EPI>
__global__ __launch_bounds__(256) void gemm_wmma_kernel(
    const bf16* __restrict__ A, const bf16* __restrict__ Bm,
    float* __restrict__ Cf, bf16* __restrict__ Cb,
    const float* __restrict__ bias, const float* __restrict__ resid,
    int M, int N, int K,
    unsigned long long sA, unsigned long long sB, unsigned long long sC) {
  __shared__ __align__(16) bf16 As[2][64][32];   // [buf][m][k]
  __shared__ __align__(16) bf16 Bs[2][64][32];   // [buf][n][k]  (transposed!)
  const int tid = threadIdx.x;
  const int lane = tid & 31, wave = tid >> 5;
  const int hi = lane >> 4, ln = lane & 15;
  const int tm = wave >> 1, tn = wave & 1;
  const int m0 = blockIdx.y * 64, n0 = blockIdx.x * 64;
  const size_t zA = (size_t)blockIdx.z * sA;
  const size_t zB = (size_t)blockIdx.z * sB;
  const size_t zC = (size_t)blockIdx.z * sC;

  // per-thread staging pointers (incremented by a constant per K-step)
  const int ar = tid >> 2, ac = (tid & 3) * 8;          // A: 8 contiguous k
  const int bkk = tid >> 3, bc8 = (tid & 7) * 8;        // B !TRANSB: 8 contiguous n
  const bf16* aSrc = A + zA + (size_t)(m0 + ar) * K + ac;
  const bf16* bSrc = TRANSB ? (Bm + zB + (size_t)(n0 + ar) * K + ac)
                            : (Bm + zB + (size_t)bkk * N + n0 + bc8);
  const size_t bStep = TRANSB ? 32 : 32 * (size_t)N;

  v8f acc0, acc1;
#pragma unroll
  for (int e = 0; e < 8; ++e) { acc0[e] = 0.0f; acc1[e] = 0.0f; }

  // prologue: stage tile 0 into buffer 0
  v4u aReg = *(const v4u*)aSrc;
  v4u bReg = *(const v4u*)bSrc;
  *(v4u*)(&As[0][ar][ac]) = aReg;
  if (TRANSB) {
    *(v4u*)(&Bs[0][ar][ac]) = bReg;
  } else {
    const bf16* tp = (const bf16*)&bReg;
#pragma unroll
    for (int j = 0; j < 8; ++j) Bs[0][bc8 + j][bkk] = tp[j];
  }
  __syncthreads();

  int buf = 0;
  for (int k0 = 0; k0 < K; k0 += 32) {
    const bool hasNext = (k0 + 32 < K);
    if (hasNext) {  // issue next tile's loads now; latency hidden by WMMAs below
      aSrc += 32;
      bSrc += bStep;
      aReg = *(const v4u*)aSrc;
      bReg = *(const v4u*)bSrc;
      __builtin_prefetch((const void*)(aSrc + 32), 0, 1);      // global_prefetch_b8
      __builtin_prefetch((const void*)(bSrc + bStep), 0, 1);
    }
    const bf16* arow = &As[buf][tm * 16 + ln][0];
    v16bf af  = cat8(*(const v8bf*)(arow + hi * 16), *(const v8bf*)(arow + hi * 16 + 8));
    v16bf bf0 = load_frag16(&Bs[buf][tn * 32 + ln][hi * 16]);
    v16bf bf1 = load_frag16(&Bs[buf][tn * 32 + 16 + ln][hi * 16]);
    acc0 = __builtin_amdgcn_wmma_f32_16x16x32_bf16(false, af, false, bf0, (short)0, acc0, false, false);
    acc1 = __builtin_amdgcn_wmma_f32_16x16x32_bf16(false, af, false, bf1, (short)0, acc1, false, false);
    if (hasNext) {  // fill the other buffer
      const int nb = buf ^ 1;
      *(v4u*)(&As[nb][ar][ac]) = aReg;
      if (TRANSB) {
        *(v4u*)(&Bs[nb][ar][ac]) = bReg;
      } else {
        const bf16* tp = (const bf16*)&bReg;
#pragma unroll
        for (int j = 0; j < 8; ++j) Bs[nb][bc8 + j][bkk] = tp[j];
      }
    }
    __syncthreads();
    buf ^= 1;
  }

#pragma unroll
  for (int sub = 0; sub < 2; ++sub) {
    v8f acc = sub ? acc1 : acc0;
    int n = n0 + tn * 32 + sub * 16 + ln;
#pragma unroll
    for (int v = 0; v < 8; ++v) {
      int m = m0 + tm * 16 + hi * 8 + v;
      size_t idx = zC + (size_t)m * N + n;
      float val = acc[v];
      if (EPI == 0) {
        Cf[idx] = val;
      } else if (EPI == 1) {
        val += bias[n];
        val = 0.5f * val * (1.0f + erff(val * 0.70710678118f));  // exact GELU
        Cb[idx] = (bf16)val;
      } else if (EPI == 2) {
        Cf[idx] = val + bias[n] + resid[idx];
      } else {
        Cf[idx] = val + resid[idx];
      }
    }
  }
}

// ---------------------------------------------------------------------------
// Fused retention: per (b,head), per 64-row S tile:
//   loop causal 32-wide t-chunks:  A = (Q K^T) * gamma^(s-t) [s>=t]  (WMMA)
//                                  Y += A V                          (WMMA)
// V chunk is stored transposed (Vt[d][t]) so its fragments are contiguous.
// ---------------------------------------------------------------------------
__global__ __launch_bounds__(256) void retention_wmma_kernel(
    const bf16* __restrict__ Qh, const bf16* __restrict__ Kh, const bf16* __restrict__ Vh,
    float* __restrict__ Y) {
  __shared__ __align__(16) bf16 Qs[64][128];  // [s][d]
  __shared__ __align__(16) bf16 Ks[32][128];  // [t][d]
  __shared__ __align__(16) bf16 Vt[128][32];  // [d][t]  (transposed!)
  __shared__ __align__(16) bf16 Sc[64][32];   // scores  [s][t]
  const int tid = threadIdx.x;
  const int lane = tid & 31, wave = tid >> 5;
  const int hi = lane >> 4, ln = lane & 15;
  const int tm = wave >> 1, tn = wave & 1;
  const int head = blockIdx.y % kHeads, b = blockIdx.y / kHeads;
  const int s0 = blockIdx.x * 64;
  const size_t base = (size_t)b * kS * kH + (size_t)head * kHD;

  // gamma = 1 - exp(linspace(log(1/32), log(1/512))[head]); decay in log space
  float tt = (float)head / (float)(kHeads - 1);
  float lgLo = logf(1.0f / 32.0f), lgHi = logf(1.0f / 512.0f);
  float gamma = 1.0f - __expf(lgLo + (lgHi - lgLo) * tt);
  float lg = logf(gamma);

  // stage Q tile (64 x 128) once
#pragma unroll
  for (int j = 0; j < 4; ++j) {
    int lin = tid + j * 256;
    int r = lin >> 4, c = (lin & 15) * 8;
    *(v4u*)(&Qs[r][c]) = *(const v4u*)(Qh + base + (size_t)(s0 + r) * kH + c);
  }

  v8f yacc[4];
#pragma unroll
  for (int c4 = 0; c4 < 4; ++c4)
#pragma unroll
    for (int e = 0; e < 8; ++e) yacc[c4][e] = 0.0f;

  const int tChunks = 2 * (blockIdx.x + 1);  // causal: t <= s0+63
  for (int kt = 0; kt < tChunks; ++kt) {
    const int t0 = kt * 32;
    __syncthreads();  // Qs visible on 1st iter; K/Vt/Sc safe to overwrite after
    {  // K chunk 32x128, row-major, contiguous
      int r = tid >> 3, c = (tid & 7) * 16;
      *(v4u*)(&Ks[r][c])     = *(const v4u*)(Kh + base + (size_t)(t0 + r) * kH + c);
      *(v4u*)(&Ks[r][c + 8]) = *(const v4u*)(Kh + base + (size_t)(t0 + r) * kH + c + 8);
    }
    {  // V chunk: coalesced row read, scatter-transpose into Vt[d][t]
      int t = tid >> 3, d0 = (tid & 7) * 16;
      v4u w0 = *(const v4u*)(Vh + base + (size_t)(t0 + t) * kH + d0);
      v4u w1 = *(const v4u*)(Vh + base + (size_t)(t0 + t) * kH + d0 + 8);
      const bf16* p0 = (const bf16*)&w0;
      const bf16* p1 = (const bf16*)&w1;
#pragma unroll
      for (int j = 0; j < 8; ++j) { Vt[d0 + j][t] = p0[j]; Vt[d0 + 8 + j][t] = p1[j]; }
    }
    if (kt + 1 < tChunks) {  // prefetch next K/V chunk
      const size_t pre = base + (size_t)(t0 + 32 + (tid >> 3)) * kH + (tid & 7) * 16;
      __builtin_prefetch((const void*)(Kh + pre), 0, 1);
      __builtin_prefetch((const void*)(Vh + pre), 0, 1);
    }
    __syncthreads();

    // ---- scores: 64x32 = Q(64x128) * K^T, each wave one 16x16 tile
    v8f acc;
#pragma unroll
    for (int e = 0; e < 8; ++e) acc[e] = 0.0f;
#pragma unroll
    for (int kk = 0; kk < 4; ++kk) {
      const bf16* qrow = &Qs[tm * 16 + ln][kk * 32];
      v16bf af  = cat8(*(const v8bf*)(qrow + hi * 16), *(const v8bf*)(qrow + hi * 16 + 8));
      v16bf bfr = load_frag16(&Ks[tn * 16 + ln][kk * 32 + hi * 16]);  // B=K^T row-contiguous
      acc = __builtin_amdgcn_wmma_f32_16x16x32_bf16(false, af, false, bfr, (short)0, acc, false, false);
    }
    // decay mask in registers, spill to LDS as bf16
#pragma unroll
    for (int v = 0; v < 8; ++v) {
      int sg = s0 + tm * 16 + hi * 8 + v;
      int tg = t0 + tn * 16 + ln;
      int dlt = sg - tg;
      float wgt = (dlt < 0) ? 0.0f : __expf(lg * (float)dlt);
      Sc[tm * 16 + hi * 8 + v][tn * 16 + ln] = (bf16)(acc[v] * wgt);
    }
    __syncthreads();

    // ---- Y(64x128) += A(64x32) * V(32x128); wave: rows tm, col group tn*64
    const bf16* srow = &Sc[tm * 16 + ln][0];
    v16bf af2 = cat8(*(const v8bf*)(srow + hi * 16), *(const v8bf*)(srow + hi * 16 + 8));
#pragma unroll
    for (int c4 = 0; c4 < 4; ++c4) {
      int col0 = (tn * 4 + c4) * 16;
      v16bf bfr = load_frag16(&Vt[col0 + ln][hi * 16]);  // contiguous thanks to transpose
      yacc[c4] = __builtin_amdgcn_wmma_f32_16x16x32_bf16(false, af2, false, bfr, (short)0, yacc[c4], false, false);
    }
  }
  __syncthreads();
#pragma unroll
  for (int c4 = 0; c4 < 4; ++c4) {
    int col = (tn * 4 + c4) * 16 + ln;
#pragma unroll
    for (int v = 0; v < 8; ++v) {
      int m = s0 + tm * 16 + hi * 8 + v;
      Y[base + (size_t)m * kH + col] = yacc[c4][v];
    }
  }
}

// ---------------------------------------------------------------------------
// Host orchestration
// ---------------------------------------------------------------------------
static void launch_gemm(hipStream_t st, const bf16* A, const bf16* B, float* Cf, bf16* Cb,
                        const float* bias, const float* resid, int M, int N, int K,
                        int transB, size_t sA, size_t sB, size_t sC, int batch, int epi) {
  dim3 grid(N / 64, M / 64, batch), block(256);
#define GO(TB, EP)                                                                     \
  gemm_wmma_kernel<TB, EP><<<grid, block, 0, st>>>(A, B, Cf, Cb, bias, resid, M, N, K, \
                                                   (unsigned long long)sA,             \
                                                   (unsigned long long)sB,             \
                                                   (unsigned long long)sC)
  if (transB)            GO(1, 0);
  else if (epi == 1)     GO(0, 1);
  else if (epi == 2)     GO(0, 2);
  else if (resid)        GO(0, 3);
  else                   GO(0, 0);
#undef GO
}

extern "C" void kernel_launch(void* const* d_in, const int* in_sizes, int n_in,
                              void* d_out, int out_size, void* d_ws, size_t ws_size,
                              hipStream_t stream) {
  (void)in_sizes; (void)n_in; (void)out_size; (void)ws_size;
  const float* X     = (const float*)d_in[0];
  const float* WQ    = (const float*)d_in[1];
  const float* WK    = (const float*)d_in[2];
  const float* WV    = (const float*)d_in[3];
  const float* WG    = (const float*)d_in[4];
  const float* WO    = (const float*)d_in[5];
  const float* gn_w  = (const float*)d_in[6];
  const float* gn_b  = (const float*)d_in[7];
  const float* ln1_w = (const float*)d_in[8];
  const float* ln1_b = (const float*)d_in[9];
  const float* ln2_w = (const float*)d_in[10];
  const float* ln2_b = (const float*)d_in[11];
  const float* fw1   = (const float*)d_in[12];
  const float* fb1   = (const float*)d_in[13];
  const float* fw2   = (const float*)d_in[14];
  const float* fb2   = (const float*)d_in[15];
  float* out = (float*)d_out;

  const size_t NBS = (size_t)kB * kS;          // 8192 rows
  char* wsb = (char*)d_ws;
  size_t off = 0;
  auto take = [&](size_t bytes) -> void* {
    void* r = wsb + off;
    off = (off + bytes + 255) & ~(size_t)255;
    return r;
  };
  float* Xcur = (float*)take(NBS * kH * sizeof(float));
  float* Qf   = (float*)take(NBS * kH * sizeof(float));
  float* Kf   = (float*)take(NBS * kH * sizeof(float));
  float* Vf   = (float*)take(NBS * kH * sizeof(float));
  float* Xres = (float*)take(NBS * kH * sizeof(float));
  bf16* Xn = (bf16*)take(NBS * kH * sizeof(bf16));
  bf16* Qh = (bf16*)take(NBS * kH * sizeof(bf16));
  bf16* Kh = (bf16*)take(NBS * kH * sizeof(bf16));
  bf16* Vh = (bf16*)take(NBS * kH * sizeof(bf16));
  bf16* F1 = (bf16*)take(NBS * kFFN * sizeof(bf16));
  bf16* WQp = (bf16*)take((size_t)kLayers * kH * kH * sizeof(bf16));
  bf16* WKp = (bf16*)take((size_t)kLayers * kH * kH * sizeof(bf16));
  bf16* WVp = (bf16*)take((size_t)kLayers * kH * kH * sizeof(bf16));
  bf16* WGp = (bf16*)take((size_t)kLayers * kH * kH * sizeof(bf16));
  bf16* WOp = (bf16*)take((size_t)kLayers * kH * kH * sizeof(bf16));
  bf16* W1p = (bf16*)take((size_t)kLayers * kH * kFFN * sizeof(bf16));
  bf16* W2p = (bf16*)take((size_t)kLayers * kFFN * kH * sizeof(bf16));
  // transient aliases (stream-ordered so these are safe)
  float* Yf = Kf;     // retention output after Kf consumed by xpos
  float* Gf = Qf;     // gate projection after Qf consumed by xpos
  bf16*  gate = Xn;   // gate elementwise output after Xn consumed
  bf16*  Hn = Qh;     // ln2 output after Qh consumed by retention
  bf16*  Xg = Xn;     // bf16 X for the Gram matrix

  const size_t nH  = NBS * kH;
  const int cvtBlocks = (int)((nH + 255) / 256);

  // ---- pack / convert weights to bf16 (deterministic every call) ----
  {
    const size_t qkvTot = (size_t)kHeads * kH * kHD;
    const int qb = (int)((qkvTot + 255) / 256);
    const size_t hh = (size_t)kH * kH, hf = (size_t)kH * kFFN;
    for (int l = 0; l < kLayers; ++l) {
      pack_qkv_kernel<<<qb, 256, 0, stream>>>(WQ + l * qkvTot, WQp + l * hh);
      pack_qkv_kernel<<<qb, 256, 0, stream>>>(WK + l * qkvTot, WKp + l * hh);
      pack_qkv_kernel<<<qb, 256, 0, stream>>>(WV + l * qkvTot, WVp + l * hh);
      convert_f32_bf16_kernel<<<(int)((hh + 255) / 256), 256, 0, stream>>>(WG + l * hh, WGp + l * hh, hh);
      convert_f32_bf16_kernel<<<(int)((hh + 255) / 256), 256, 0, stream>>>(WO + l * hh, WOp + l * hh, hh);
      convert_f32_bf16_kernel<<<(int)((hf + 255) / 256), 256, 0, stream>>>(fw1 + l * hf, W1p + l * hf, hf);
      convert_f32_bf16_kernel<<<(int)((hf + 255) / 256), 256, 0, stream>>>(fw2 + l * hf, W2p + l * hf, hf);
    }
  }
  copy_f32_kernel<<<cvtBlocks, 256, 0, stream>>>(X, Xcur, nH);

  const size_t xposTot = (size_t)kB * kS * kHeads * (kHD / 2);
  for (int l = 0; l < kLayers; ++l) {
    const size_t lhh = (size_t)l * kH * kH, lhf = (size_t)l * kH * kFFN;
    // LN1 -> bf16
    layernorm_bf16_kernel<<<(int)NBS, 256, 0, stream>>>(Xcur, ln1_w + l * kH, ln1_b + l * kH, Xn);
    // Q/K/V projections (fp32 out)
    launch_gemm(stream, Xn, WQp + lhh, Qf, nullptr, nullptr, nullptr, (int)NBS, kH, kH, 0, 0, 0, 0, 1, 0);
    launch_gemm(stream, Xn, WKp + lhh, Kf, nullptr, nullptr, nullptr, (int)NBS, kH, kH, 0, 0, 0, 0, 1, 0);
    launch_gemm(stream, Xn, WVp + lhh, Vf, nullptr, nullptr, nullptr, (int)NBS, kH, kH, 0, 0, 0, 0, 1, 0);
    // xPos rotary -> bf16 Q/K; plain convert V
    xpos_kernel<<<(int)(xposTot / 256), 256, 0, stream>>>(Qf, Kf, Qh, Kh);
    convert_f32_bf16_kernel<<<cvtBlocks, 256, 0, stream>>>(Vf, Vh, nH);
    // fused decay retention: Y = (QK^T . D) V
    retention_wmma_kernel<<<dim3(kS / 64, kB * kHeads), 256, 0, stream>>>(Qh, Kh, Vh, Yf);
    // gate projection G = Xn @ WG (reuses Qf storage)
    launch_gemm(stream, Xn, WGp + lhh, Gf, nullptr, nullptr, nullptr, (int)NBS, kH, kH, 0, 0, 0, 0, 1, 0);
    // groupnorm + SiLU gate -> bf16 (reuses Xn storage)
    gate_kernel<<<(int)(NBS * kHeads), 128, 0, stream>>>(Yf, Gf, gn_w + l * kH, gn_b + l * kH, gate);
    // output projection + residual: Xres = gate @ WO + Xcur
    launch_gemm(stream, gate, WOp + lhh, Xres, nullptr, nullptr, Xcur, (int)NBS, kH, kH, 0, 0, 0, 0, 1, 0);
    // LN2 -> bf16 (reuses Qh storage)
    layernorm_bf16_kernel<<<(int)NBS, 256, 0, stream>>>(Xres, ln2_w + l * kH, ln2_b + l * kH, Hn);
    // FFN1: gelu(Hn @ W1 + b1) -> bf16
    launch_gemm(stream, Hn, W1p + lhf, nullptr, F1, fb1 + l * kFFN, nullptr, (int)NBS, kFFN, kH, 0, 0, 0, 0, 1, 1);
    // FFN2: Xcur = F1 @ W2 + b2 + Xres
    launch_gemm(stream, F1, W2p + lhf, Xcur, nullptr, fb2 + l * kH, Xres, (int)NBS, kH, kFFN, 0, 0, 0, 0, 1, 2);
  }

  // outputs: (X, X @ X^T per batch)
  copy_f32_kernel<<<cvtBlocks, 256, 0, stream>>>(Xcur, out, nH);
  convert_f32_bf16_kernel<<<cvtBlocks, 256, 0, stream>>>(Xcur, Xg, nH);
  launch_gemm(stream, Xg, Xg, out + nH, nullptr, nullptr, nullptr, kS, kS, kH, /*transB=*/1,
              (size_t)kS * kH, (size_t)kS * kH, (size_t)kS * kS, kB, 0);
}
